// NMT_30374008717729
// MI455X (gfx1250) — matrix operand order repaired and checked by reading, not compiled, not run on hardware
//
#include <hip/hip_runtime.h>

#define TLEN  50
#define BATCH 128
#define HDIM  1024
#define EWDIM 1024
#define TWDIM 512
#define ODIM  512
#define TB    (TLEN*BATCH)

typedef __attribute__((ext_vector_type(16))) _Float16 v16h;
typedef __attribute__((ext_vector_type(8)))  _Float16 v8h;
typedef __attribute__((ext_vector_type(8)))  float    v8f;

union HV16 { v16h v; v8h h[2]; };

// ---------------------------------------------------------------------------
// WMMA GEMM: C[M,N](f32) = A[M,K](f16, row stride lda) * BT[N,K](f16) + bias[N]
// BT is B transposed (row n = column n of B) so each lane reads contiguous K.
// Block = 256 threads = 8 waves arranged 2(M) x 4(N); each wave computes a
// 16x64 strip of C with 4 accumulators, reusing its A fragment 4x per k-chunk.
// All 5 operand loads of a k-chunk are issued before the 4 WMMAs so the
// scheduler can use partial s_wait_loadcnt and overlap loads with matrix ops.
// grid = (N/256, M/32).  Requires N%256==0, M%32==0, K%32==0 (all call sites).
// act: 0 = none, 1 = tanh.
// ---------------------------------------------------------------------------
__global__ __launch_bounds__(256)
void nmt_gemm_f16(const _Float16* __restrict__ A, int lda,
                  const _Float16* __restrict__ BT,
                  float* __restrict__ C,
                  const float* __restrict__ bias,
                  int N, int K, int act)
{
  const int lane = threadIdx.x & 31;
  const int wave = threadIdx.x >> 5;
  const int mw   = wave >> 2;          // 0..1
  const int nw   = wave & 3;           // 0..3
  const int hseg = lane >> 4;          // 0 or 1 (lane group)
  const int l16  = lane & 15;
  const int row0 = blockIdx.y * 32 + mw * 16;
  const int col0 = blockIdx.x * 256 + nw * 64;

  const _Float16* arow = A  + (size_t)(row0 + l16) * lda + hseg * 16;
  const _Float16* b0   = BT + (size_t)(col0 + l16) * K   + hseg * 16;
  const _Float16* b1   = b0 + (size_t)16 * K;
  const _Float16* b2   = b0 + (size_t)32 * K;
  const _Float16* b3   = b0 + (size_t)48 * K;

  v8f acc0 = {}, acc1 = {}, acc2 = {}, acc3 = {};
  for (int k = 0; k < K; k += 32) {
    if (k + 512 < K) {                  // stream-prefetch the B operand
      __builtin_prefetch(b0 + k + 512, 0, 3);
      __builtin_prefetch(b2 + k + 512, 0, 3);
    }
    // issue all operand loads for this k-chunk up front
    HV16 a, bv0, bv1, bv2, bv3;
    a.h[0] = *(const v8h*)(arow + k);        // K run [k+hseg*16,   +8)
    a.h[1] = *(const v8h*)(arow + k + 8);    // K run [k+8+hseg*16, +8)
    bv0.v  = *(const v16h*)(b0 + k);
    bv1.v  = *(const v16h*)(b1 + k);
    bv2.v  = *(const v16h*)(b2 + k);
    bv3.v  = *(const v16h*)(b3 + k);
    acc0 = __builtin_amdgcn_wmma_f32_16x16x32_f16(false, a.v, false, bv0.v,
                                                  (short)0, acc0, false, false);
    acc1 = __builtin_amdgcn_wmma_f32_16x16x32_f16(false, a.v, false, bv1.v,
                                                  (short)0, acc1, false, false);
    acc2 = __builtin_amdgcn_wmma_f32_16x16x32_f16(false, a.v, false, bv2.v,
                                                  (short)0, acc2, false, false);
    acc3 = __builtin_amdgcn_wmma_f32_16x16x32_f16(false, a.v, false, bv3.v,
                                                  (short)0, acc3, false, false);
  }

#pragma unroll
  for (int s = 0; s < 4; ++s) {
    const v8f acc = (s == 0) ? acc0 : (s == 1) ? acc1 : (s == 2) ? acc2 : acc3;
    const int   col = col0 + s * 16 + l16;
    const float bv  = bias ? bias[col] : 0.0f;
#pragma unroll
    for (int r = 0; r < 8; ++r) {
      float v = acc[r] + bv;
      if (act == 1) v = tanhf(v);
      const int row = row0 + r + hseg * 8;
      C[(size_t)row * N + col] = v;
    }
  }
}

// ---------------------------------------------------------------------------
// Recurrent-step WMMA GEMM (M=128, K=1024, no bias/act): cg = c_prev @ W_cif.
// A (c_h) is tiny and reused by every N-tile, so each block stages its 32-row
// A slice through LDS with gfx1250 async global->LDS DMA (two 32KB K-halves),
// synchronized via s_wait_asynccnt + workgroup barrier; WMMA then reads A
// from LDS (ds_load_b128) while the B weights stream from global/L2.
// grid = (N/256, M/32); lda must equal 1024.
// ---------------------------------------------------------------------------
#define CGK  1024
#define CGKH 512                          // K half staged per round

__global__ __launch_bounds__(256)
void nmt_gemm_cg(const _Float16* __restrict__ A,
                 const _Float16* __restrict__ BT,
                 float* __restrict__ C, int N)
{
  __shared__ _Float16 As[32 * CGKH];      // 32 KB stage
  const int tid  = threadIdx.x;
  const int lane = tid & 31;
  const int wave = tid >> 5;
  const int mw   = wave >> 2;
  const int nw   = wave & 3;
  const int hseg = lane >> 4;
  const int l16  = lane & 15;
  const int row0 = blockIdx.y * 32;
  const int col0 = blockIdx.x * 256 + nw * 64;

  const _Float16* b0 = BT + (size_t)(col0 + l16) * CGK + hseg * 16;
  const _Float16* b1 = b0 + (size_t)16 * CGK;
  const _Float16* b2 = b0 + (size_t)32 * CGK;
  const _Float16* b3 = b0 + (size_t)48 * CGK;

  const unsigned lbase = (unsigned)(size_t)(void*)As;   // workgroup-relative LDS offset
  v8f acc0 = {}, acc1 = {}, acc2 = {}, acc3 = {};

  for (int kh = 0; kh < 2; ++kh) {
    if (kh) __syncthreads();              // prior reads done before overwrite
    // async-copy 32 x 512 f16 (32KB): 8 x b128 per thread
#pragma unroll
    for (int it = 0; it < 8; ++it) {
      const int idx = (it * 256 + tid) * 8;          // half index in stage
      const int r   = idx >> 9;                      // 512 halfs per row
      const int kk  = idx & 511;
      const _Float16* g = A + (size_t)(row0 + r) * CGK + kh * CGKH + kk;
      const unsigned  l = lbase + (unsigned)idx * 2;
      asm volatile("global_load_async_to_lds_b128 %0, %1, off"
                   :: "v"(l), "v"(g) : "memory");
    }
    asm volatile("s_wait_asynccnt 0x0" ::: "memory");
    __syncthreads();

    const _Float16* as_row = As + (size_t)(mw * 16 + l16) * CGKH + hseg * 16;
    const int kbase = kh * CGKH;
    for (int k = 0; k < CGKH; k += 32) {
      // all operand loads first (A from LDS, 4x B from global), then 4 WMMAs
      HV16 a, bv0, bv1, bv2, bv3;
      a.h[0] = *(const v8h*)(as_row + k);
      a.h[1] = *(const v8h*)(as_row + k + 8);
      bv0.v  = *(const v16h*)(b0 + kbase + k);
      bv1.v  = *(const v16h*)(b1 + kbase + k);
      bv2.v  = *(const v16h*)(b2 + kbase + k);
      bv3.v  = *(const v16h*)(b3 + kbase + k);
      acc0 = __builtin_amdgcn_wmma_f32_16x16x32_f16(false, a.v, false, bv0.v,
                                                    (short)0, acc0, false, false);
      acc1 = __builtin_amdgcn_wmma_f32_16x16x32_f16(false, a.v, false, bv1.v,
                                                    (short)0, acc1, false, false);
      acc2 = __builtin_amdgcn_wmma_f32_16x16x32_f16(false, a.v, false, bv2.v,
                                                    (short)0, acc2, false, false);
      acc3 = __builtin_amdgcn_wmma_f32_16x16x32_f16(false, a.v, false, bv3.v,
                                                    (short)0, acc3, false, false);
    }
  }

#pragma unroll
  for (int s = 0; s < 4; ++s) {
    const v8f acc = (s == 0) ? acc0 : (s == 1) ? acc1 : (s == 2) ? acc2 : acc3;
    const int col = col0 + s * 16 + l16;
#pragma unroll
    for (int r = 0; r < 8; ++r) {
      const int row = row0 + r + hseg * 8;
      C[(size_t)row * N + col] = acc[r];
    }
  }
}

// ---------------------------------------------------------------------------
// Weight convert + transpose: W[K,N] f32 -> WT[N,K] f16
// ---------------------------------------------------------------------------
__global__ void nmt_wt(const float* __restrict__ W, _Float16* __restrict__ WT,
                       int K, int N)
{
  int i = blockIdx.x * blockDim.x + threadIdx.x;
  if (i >= K * N) return;
  int k = i / N, n = i % N;
  WT[(size_t)n * K + k] = (_Float16)W[i];
}

__global__ void nmt_cvt(const float* __restrict__ x, _Float16* __restrict__ y, int n)
{
  int i = blockIdx.x * blockDim.x + threadIdx.x;
  if (i < n) y[i] = (_Float16)x[i];
}

__global__ void nmt_zero_state(float* c, _Float16* ch, int n)
{
  int i = blockIdx.x * blockDim.x + threadIdx.x;
  if (i < n) { c[i] = 0.0f; ch[i] = (_Float16)0.0f; }
}

__global__ void nmt_copy_state(const float* __restrict__ s, float* c, _Float16* ch, int n)
{
  int i = blockIdx.x * blockDim.x + threadIdx.x;
  if (i < n) { float v = s[i]; c[i] = v; ch[i] = (_Float16)v; }
}

// ---------------------------------------------------------------------------
// Embedding gather: rows of emb[V,E] selected by tok -> fp32 (optional) + f16
// ---------------------------------------------------------------------------
__global__ void nmt_gather(const int* __restrict__ tok,
                           const float* __restrict__ emb,
                           float* __restrict__ of, _Float16* __restrict__ oh,
                           int rows, int E)
{
  size_t i = (size_t)blockIdx.x * blockDim.x + threadIdx.x;
  if (i >= (size_t)rows * E) return;
  int r = (int)(i / E);
  int k = (int)(i % E);
  float v = emb[(size_t)tok[r] * E + k];
  if (of) of[i] = v;
  oh[i] = (_Float16)v;
}

// ---------------------------------------------------------------------------
// Fused RAN cell elementwise (one timestep):
//   g  = sigmoid(xif + cg) ; i=g[:H], f=g[H:]
//   cn = i*xc + f*c_prev ; cn = m*cn + (1-m)*c_prev
//   h  = tanh(cn) (+ residual)
// Writes new c (fp32 + f16 for next step's WMMA GEMM) and h (fp32/f16).
// ---------------------------------------------------------------------------
__global__ __launch_bounds__(256)
void nmt_ran_step(const float* __restrict__ xif, const float* __restrict__ xc,
                  const float* __restrict__ cg,  const float* __restrict__ m,
                  const float* __restrict__ xres,
                  float* __restrict__ c, _Float16* __restrict__ ch,
                  float* __restrict__ hout, _Float16* __restrict__ hh,
                  int hh_stride, int B, int H)
{
  int i = blockIdx.x * blockDim.x + threadIdx.x;
  if (i >= B * H) return;
  int b = i / H, k = i % H;
  float gi = xif[(size_t)b*2*H + k]     + cg[(size_t)b*2*H + k];
  float gf = xif[(size_t)b*2*H + H + k] + cg[(size_t)b*2*H + H + k];
  gi = 1.0f / (1.0f + expf(-gi));
  gf = 1.0f / (1.0f + expf(-gf));
  float cp = c[i];
  float cn = gi * xc[i] + gf * cp;
  if (m) { float mm = m[b]; cn = mm*cn + (1.0f-mm)*cp; }
  float h = tanhf(cn);
  if (xres) h += xres[i];
  c[i]  = cn;
  ch[i] = (_Float16)cn;
  if (hout) hout[i] = h;
  if (hh)   hh[(size_t)b * hh_stride + k] = (_Float16)h;
}

// ---------------------------------------------------------------------------
// Masked mean over time: mean[b,h] = sum_t x[t,b,h]*m[t,b] / sum_t m[t,b]
// ---------------------------------------------------------------------------
__global__ void nmt_mean(const float* __restrict__ x, const float* __restrict__ m,
                         float* __restrict__ mf, _Float16* __restrict__ mh,
                         int B, int H)
{
  int i = blockIdx.x * blockDim.x + threadIdx.x;
  if (i >= B * H) return;
  int b = i / H;
  float acc = 0.0f, ms = 0.0f;
  for (int t = 0; t < TLEN; ++t) {
    float mm = m[t*B + b];
    acc += x[(size_t)t*B*H + i] * mm;
    ms  += mm;
  }
  float v = acc / ms;
  mf[i] = v;
  mh[i] = (_Float16)v;
}

// ---------------------------------------------------------------------------
// Attention scores: e[ts,t,b] = exp(tanh(sa[t,b,:]+uh[ts,b,:]).w_a1 + b_a1)*m[ts,b]
// grid (t, b), 256 threads, H must be 1024 (4 elems / thread).
// ---------------------------------------------------------------------------
__global__ __launch_bounds__(256)
void nmt_attn_score(const float* __restrict__ sa, const float* __restrict__ uh,
                    const float* __restrict__ w_a1, const float* __restrict__ b_a1,
                    const float* __restrict__ srcs_m, float* __restrict__ e,
                    int Tt, int B, int H)
{
  int t = blockIdx.x, b = blockIdx.y;
  __shared__ float red[256];
  float sav[4], wv[4];
  size_t base = ((size_t)t*B + b) * H;
#pragma unroll
  for (int j = 0; j < 4; ++j) {
    int idx = threadIdx.x + j*256;
    sav[j] = sa[base + idx];
    wv[j]  = w_a1[idx];
  }
  float ba = b_a1[0];
  for (int ts = 0; ts < TLEN; ++ts) {
    const float* u = uh + ((size_t)ts*B + b) * H;
    float p = 0.0f;
#pragma unroll
    for (int j = 0; j < 4; ++j) {
      int idx = threadIdx.x + j*256;
      p += tanhf(sav[j] + u[idx]) * wv[j];
    }
    red[threadIdx.x] = p;
    __syncthreads();
    for (int s = 128; s > 0; s >>= 1) {
      if (threadIdx.x < s) red[threadIdx.x] += red[threadIdx.x + s];
      __syncthreads();
    }
    if (threadIdx.x == 0)
      e[((size_t)ts*Tt + t)*B + b] = expf(red[0] + ba) * srcs_m[ts*B + b];
    __syncthreads();
  }
}

__global__ void nmt_attn_norm(float* __restrict__ e, int Tt, int B)
{
  int i = blockIdx.x * blockDim.x + threadIdx.x;
  if (i >= Tt * B) return;
  int t = i / B, b = i % B;
  float s = 0.0f;
  for (int ts = 0; ts < TLEN; ++ts) s += e[((size_t)ts*Tt + t)*B + b];
  float inv = 1.0f / s;
  for (int ts = 0; ts < TLEN; ++ts) e[((size_t)ts*Tt + t)*B + b] *= inv;
}

// attends[t,b,:] = sum_ts e[ts,t,b] * x_es[ts,b,:] -> f16 into concat[..., H:2H]
__global__ __launch_bounds__(256)
void nmt_attn_apply(const float* __restrict__ e, const float* __restrict__ x_es,
                    _Float16* __restrict__ cat, int Tt, int B, int H)
{
  int t = blockIdx.x, b = blockIdx.y;
  float acc[4] = {0.f, 0.f, 0.f, 0.f};
  for (int ts = 0; ts < TLEN; ++ts) {
    float ev = e[((size_t)ts*Tt + t)*B + b];
    const float* x = x_es + ((size_t)ts*B + b) * H;
#pragma unroll
    for (int j = 0; j < 4; ++j) acc[j] += ev * x[threadIdx.x + j*256];
  }
  _Float16* dst = cat + ((size_t)t*B + b) * (2*H) + H;
#pragma unroll
  for (int j = 0; j < 4; ++j) dst[threadIdx.x + j*256] = (_Float16)acc[j];
}

// out[t,b,o] = max(lg[t*B+b, 2o], lg[t*B+b, 2o+1]) * trgs_m[t,b]
__global__ void nmt_logit(const float* __restrict__ lg, const float* __restrict__ m,
                          float* __restrict__ out, int n, int O)
{
  int i = blockIdx.x * blockDim.x + threadIdx.x;
  if (i >= n) return;
  int o = i % O, tb = i / O;
  float a = lg[(size_t)tb*2*O + 2*o];
  float b = lg[(size_t)tb*2*O + 2*o + 1];
  out[i] = fmaxf(a, b) * m[tb];
}

// ---------------------------------------------------------------------------

extern "C" void kernel_launch(void* const* d_in, const int* in_sizes, int n_in,
                              void* d_out, int out_size, void* d_ws, size_t ws_size,
                              hipStream_t stream)
{
  (void)in_sizes; (void)n_in; (void)out_size; (void)ws_size;

  const int*   srcs    = (const int*)  d_in[0];
  const int*   trgs    = (const int*)  d_in[1];
  const float* srcs_m  = (const float*)d_in[2];
  const float* trgs_m  = (const float*)d_in[3];
  const float* src_emb = (const float*)d_in[4];
  const float* trg_emb = (const float*)d_in[5];
  const float* ef_Wxif = (const float*)d_in[6];
  const float* ef_Wcif = (const float*)d_in[7];
  const float* ef_bif  = (const float*)d_in[8];
  const float* ef_Wxc  = (const float*)d_in[9];
  const float* ef_bc   = (const float*)d_in[10];
  const float* eb_Wxif = (const float*)d_in[11];
  const float* eb_Wcif = (const float*)d_in[12];
  const float* eb_bif  = (const float*)d_in[13];
  const float* eb_Wxc  = (const float*)d_in[14];
  const float* eb_bc   = (const float*)d_in[15];
  const float* r1_Wxif = (const float*)d_in[16];
  const float* r1_Wcif = (const float*)d_in[17];
  const float* r1_bif  = (const float*)d_in[18];
  const float* r1_Wxc  = (const float*)d_in[19];
  const float* r1_bc   = (const float*)d_in[20];
  const float* r2_Wxif = (const float*)d_in[21];
  const float* r2_Wcif = (const float*)d_in[22];
  const float* r2_bif  = (const float*)d_in[23];
  const float* r2_Wxc  = (const float*)d_in[24];
  const float* r2_bc   = (const float*)d_in[25];
  const float* W_sinit = (const float*)d_in[26];
  const float* b_sinit = (const float*)d_in[27];
  const float* W_ha    = (const float*)d_in[28];
  const float* b_ha    = (const float*)d_in[29];
  const float* W_sa    = (const float*)d_in[30];
  const float* b_sa    = (const float*)d_in[31];
  const float* w_a1    = (const float*)d_in[32];
  const float* b_a1    = (const float*)d_in[33];
  const float* W_ls    = (const float*)d_in[34];
  const float* b_ls    = (const float*)d_in[35];

  // ---- workspace arena ----
  char* base = (char*)d_ws;
  size_t off = 0;
  auto alloc  = [&](size_t bytes) -> void* {
    off = (off + 255) & ~(size_t)255;
    void* p = base + off;
    off += bytes;
    return p;
  };
  auto allocF = [&](size_t n) { return (float*)   alloc(n * sizeof(float)); };
  auto allocH = [&](size_t n) { return (_Float16*)alloc(n * sizeof(_Float16)); };

  // f16 transposed weights
  _Float16* wt_ef_xif = allocH((size_t)2048*1024);
  _Float16* wt_ef_cif = allocH((size_t)2048*1024);
  _Float16* wt_ef_xc  = allocH((size_t)1024*1024);
  _Float16* wt_eb_xif = allocH((size_t)2048*1024);
  _Float16* wt_eb_cif = allocH((size_t)2048*1024);
  _Float16* wt_eb_xc  = allocH((size_t)1024*1024);
  _Float16* wt_r1_xif = allocH((size_t)2048*512);
  _Float16* wt_r1_cif = allocH((size_t)2048*1024);
  _Float16* wt_r1_xc  = allocH((size_t)1024*512);
  _Float16* wt_r2_xif = allocH((size_t)2048*2048);
  _Float16* wt_r2_cif = allocH((size_t)2048*1024);
  _Float16* wt_r2_xc  = allocH((size_t)1024*2048);
  _Float16* wt_sinit  = allocH((size_t)1024*1024);
  _Float16* wt_ha     = allocH((size_t)1024*1024);
  _Float16* wt_sa     = allocH((size_t)1024*1024);
  _Float16* wt_ls     = allocH((size_t)1024*1024);

  // activations
  float*    bufA_f  = allocF((size_t)TB*HDIM);        // x_es fp32 (ping)
  _Float16* bufA_h  = allocH((size_t)TB*HDIM);
  float*    bufB_f  = allocF((size_t)TB*HDIM);        // pong
  _Float16* bufB_h  = allocH((size_t)TB*HDIM);
  float*    xif_buf = allocF((size_t)TB*2*HDIM);      // batched input proj (if gates)
  float*    xc_buf  = allocF((size_t)TB*HDIM);        // batched input proj (cand)
  float*    cg_buf  = allocF((size_t)BATCH*2*HDIM);   // per-step recurrent GEMM out
  float*    c_f     = allocF((size_t)BATCH*HDIM);     // RAN carry
  _Float16* c_h     = allocH((size_t)BATCH*HDIM);
  float*    mean_f  = allocF((size_t)BATCH*HDIM);
  _Float16* mean_h  = allocH((size_t)BATCH*HDIM);
  float*    s0_f    = allocF((size_t)BATCH*HDIM);
  _Float16* s0_h    = allocH((size_t)BATCH*HDIM);
  float*    uh_buf  = allocF((size_t)TB*HDIM);
  _Float16* yse_h   = allocH((size_t)TB*TWDIM);
  float*    e_buf   = allocF((size_t)TLEN*TLEN*BATCH);
  _Float16* cat_h   = allocH((size_t)TB*2*HDIM);      // [h | attend] f16
  _Float16* s16     = allocH((size_t)TB*HDIM);        // ran2 outputs f16
  float*    big_f   = allocF((size_t)TB*HDIM);        // sa, then logits (reused)

  // ---- launch helpers ----
  auto gemm = [&](const _Float16* A, int lda, const _Float16* BT, float* C,
                  const float* bias, int M, int N, int K, int act) {
    dim3 grid(N / 256, M / 32);
    nmt_gemm_f16<<<grid, 256, 0, stream>>>(A, lda, BT, C, bias, N, K, act);
  };
  auto wtc = [&](const float* W, _Float16* WT, int K, int N) {
    int n = K * N;
    nmt_wt<<<(n + 255) / 256, 256, 0, stream>>>(W, WT, K, N);
  };

  // ---- convert weights once (f32 -> f16 transposed) ----
  wtc(ef_Wxif, wt_ef_xif, 1024, 2048);
  wtc(ef_Wcif, wt_ef_cif, 1024, 2048);
  wtc(ef_Wxc,  wt_ef_xc,  1024, 1024);
  wtc(eb_Wxif, wt_eb_xif, 1024, 2048);
  wtc(eb_Wcif, wt_eb_cif, 1024, 2048);
  wtc(eb_Wxc,  wt_eb_xc,  1024, 1024);
  wtc(r1_Wxif, wt_r1_xif,  512, 2048);
  wtc(r1_Wcif, wt_r1_cif, 1024, 2048);
  wtc(r1_Wxc,  wt_r1_xc,   512, 1024);
  wtc(r2_Wxif, wt_r2_xif, 2048, 2048);
  wtc(r2_Wcif, wt_r2_cif, 1024, 2048);
  wtc(r2_Wxc,  wt_r2_xc,  2048, 1024);
  wtc(W_sinit, wt_sinit,  1024, 1024);
  wtc(W_ha,    wt_ha,     1024, 1024);
  wtc(W_sa,    wt_sa,     1024, 1024);
  wtc(W_ls,    wt_ls,     1024, 1024);

  // ---- embedding gathers ----
  {
    size_t n = (size_t)TB * EWDIM;
    nmt_gather<<<(unsigned)((n + 255) / 256), 256, 0, stream>>>(
        srcs, src_emb, bufA_f, bufA_h, TB, EWDIM);
  }
  {
    size_t n = (size_t)TB * TWDIM;
    nmt_gather<<<(unsigned)((n + 255) / 256), 256, 0, stream>>>(
        trgs, trg_emb, nullptr, yse_h, TB, TWDIM);
  }

  const int BH = BATCH * HDIM;
  const int BHB = (BH + 255) / 256;

  // Generic RAN scan: hoisted input projections + per-step (WMMA GEMM + fused elementwise)
  auto scan = [&](const _Float16* in_h, int din, const float* in_res,
                  const _Float16* WxifT, const float* bif,
                  const _Float16* WxcT, const float* bc,
                  const _Float16* WcifT, const float* mask,
                  float* hseq_f, _Float16* hseq_h, int hstride,
                  bool reverse, bool residual) {
    gemm(in_h, din, WxifT, xif_buf, bif, TB, 2*HDIM, din, 0);
    gemm(in_h, din, WxcT,  xc_buf,  bc,  TB, HDIM,   din, 0);
    for (int s = 0; s < TLEN; ++s) {
      int t = reverse ? (TLEN - 1 - s) : s;
      // recurrent GEMM with async LDS-staged A (M=128, N=2048, K=1024)
      {
        dim3 grid((2*HDIM) / 256, BATCH / 32);
        nmt_gemm_cg<<<grid, 256, 0, stream>>>(c_h, WcifT, cg_buf, 2*HDIM);
      }
      nmt_ran_step<<<BHB, 256, 0, stream>>>(
          xif_buf + (size_t)t*BATCH*2*HDIM,
          xc_buf  + (size_t)t*BATCH*HDIM,
          cg_buf,
          mask + (size_t)t*BATCH,
          residual ? (in_res + (size_t)t*BATCH*HDIM) : nullptr,
          c_f, c_h,
          hseq_f ? hseq_f + (size_t)t*BATCH*HDIM : nullptr,
          hseq_h + (size_t)t*BATCH*hstride, hstride,
          BATCH, HDIM);
    }
  };

  // ---- encoder: 2 layers of (forward scan, backward scan), carry c chained ----
  nmt_zero_state<<<BHB, 256, 0, stream>>>(c_f, c_h, BH);
  for (int layer = 0; layer < 2; ++layer) {
    scan(bufA_h, HDIM, bufA_f, wt_ef_xif, ef_bif, wt_ef_xc, ef_bc, wt_ef_cif,
         srcs_m, bufB_f, bufB_h, HDIM, false, true);
    scan(bufB_h, HDIM, bufB_f, wt_eb_xif, eb_bif, wt_eb_xc, eb_bc, wt_eb_cif,
         srcs_m, bufA_f, bufA_h, HDIM, true, true);
  }
  // final x_es lives in bufA (fp32 + f16)

  // ---- s0 = tanh(mean_enc @ W_sinit + b_sinit) ----
  nmt_mean<<<BHB, 256, 0, stream>>>(bufA_f, srcs_m, mean_f, mean_h, BATCH, HDIM);
  gemm(mean_h, HDIM, wt_sinit, s0_f, b_sinit, BATCH, HDIM, HDIM, 1);
  nmt_cvt<<<BHB, 256, 0, stream>>>(s0_f, s0_h, BH);

  // ---- uh = x_es @ W_ha + b_ha ----
  gemm(bufA_h, HDIM, wt_ha, uh_buf, b_ha, TB, HDIM, HDIM, 0);

  // ---- decoder scan 1 (ran1, no residual); h written f16 into concat[:,0:H] ----
  nmt_copy_state<<<BHB, 256, 0, stream>>>(s0_f, c_f, c_h, BH);
  scan(yse_h, TWDIM, nullptr, wt_r1_xif, r1_bif, wt_r1_xc, r1_bc, wt_r1_cif,
       trgs_m, nullptr, cat_h, 2*HDIM, false, false);

  // ---- attention (batched over all target steps) ----
  gemm(cat_h, 2*HDIM, wt_sa, big_f, b_sa, TB, HDIM, HDIM, 0);   // sa = hs @ W_sa
  {
    dim3 gtb(TLEN, BATCH);
    nmt_attn_score<<<gtb, 256, 0, stream>>>(big_f, uh_buf, w_a1, b_a1, srcs_m,
                                            e_buf, TLEN, BATCH, HDIM);
    nmt_attn_norm<<<(TLEN*BATCH + 255) / 256, 256, 0, stream>>>(e_buf, TLEN, BATCH);
    nmt_attn_apply<<<gtb, 256, 0, stream>>>(e_buf, bufA_f, cat_h, TLEN, BATCH, HDIM);
  }

  // ---- decoder scan 2 (ran2 over concat(h, attend), K=2048, no residual) ----
  nmt_copy_state<<<BHB, 256, 0, stream>>>(s0_f, c_f, c_h, BH);
  scan(cat_h, 2*HDIM, nullptr, wt_r2_xif, r2_bif, wt_r2_xc, r2_bc, wt_r2_cif,
       trgs_m, nullptr, s16, HDIM, false, false);

  // ---- logits: s @ W_ls + b_ls, pairwise max, mask ----
  gemm(s16, HDIM, wt_ls, big_f, b_ls, TB, 2*ODIM, HDIM, 0);
  {
    int n = TLEN * BATCH * ODIM;
    nmt_logit<<<(n + 255) / 256, 256, 0, stream>>>(big_f, trgs_m, (float*)d_out,
                                                   n, ODIM);
  }
}
